// CrossAttentionFusion_2439541424447
// MI455X (gfx1250) — matrix operand order repaired and checked by reading, not compiled, not run on hardware
//
#include <hip/hip_runtime.h>

// Problem constants (from reference): B=2, C=Co=256, T=8, H=28, W=28
#define B_  2
#define CO  256
#define N_  6272          // 8*28*28, divisible by 32
#define NWAVES 8
#define KPAD 264          // 256 + 8 bf16 -> 528B row, bank step 4
#define VPAD 40           // 32 + 8 bf16  -> 80B row,  bank step 20

typedef __attribute__((ext_vector_type(16))) __bf16 v16bf;
typedef __attribute__((ext_vector_type(8)))  __bf16 v8bf;
typedef __attribute__((ext_vector_type(8)))  float  v8f;
typedef __attribute__((ext_vector_type(4)))  float  v4f;

__device__ __forceinline__ v16bf make_a_frag(v8bf lo, v8bf hi) {
  return __builtin_shufflevector(lo, hi, 0,1,2,3,4,5,6,7,8,9,10,11,12,13,14,15);
}

// Async global->LDS 16B copy (GVS mode: SGPR base + 32-bit per-lane offset).
// Tracked by ASYNCcnt; LDS byte address is the low 32 bits of the generic ptr.
__device__ __forceinline__ void async_b128(unsigned lds_addr, unsigned voff,
                                           const void* sbase) {
  asm volatile("global_load_async_to_lds_b128 %0, %1, %2"
               :: "v"(lds_addr), "v"(voff), "s"(sbase) : "memory");
}
__device__ __forceinline__ void wait_asynccnt0() {
  asm volatile("s_wait_asynccnt 0x0" ::: "memory");
}

// ---------------------------------------------------------------------------
// Projection for Q and K: out[b][n][o] = sum_c in[b][c][n]*w[o][c] + bias[o]
// out is bf16 [B][N][CO].
// ---------------------------------------------------------------------------
__global__ __launch_bounds__(128)
void proj_qk_kernel(const float* __restrict__ in,    // [B][CO][N]
                    const float* __restrict__ w,     // [CO][CO]
                    const float* __restrict__ bias,  // [CO]
                    __bf16* __restrict__ out)        // [B][N][CO]
{
  __shared__ __align__(16) __bf16 lds_a[16][32];     // [n][c] bf16
  const int tid  = threadIdx.x;
  const int wave = tid >> 5;
  const int lane = tid & 31;
  const int lrow = lane & 15;
  const int lhi  = lane >> 4;
  const int b  = blockIdx.z;
  const int nb = blockIdx.y * 16;
  const int ob = blockIdx.x * 64 + wave * 16;

  const float* inb = in + (size_t)b * CO * N_;
  v8f acc = {};

  for (int cb = 0; cb < CO; cb += 32) {
    {
      int c  = tid >> 2;        // 0..31
      int n4 = (tid & 3) * 4;   // 0,4,8,12
      v4f xv = *(const v4f*)(inb + (size_t)(cb + c) * N_ + nb + n4);
      lds_a[n4 + 0][c] = (__bf16)xv.x;
      lds_a[n4 + 1][c] = (__bf16)xv.y;
      lds_a[n4 + 2][c] = (__bf16)xv.z;
      lds_a[n4 + 3][c] = (__bf16)xv.w;
    }
    __syncthreads();
    v8bf alo = *(const v8bf*)(&lds_a[lrow][lhi * 16]);
    v8bf ahi = *(const v8bf*)(&lds_a[lrow][lhi * 16 + 8]);
    v16bf afrag = make_a_frag(alo, ahi);
    const float* wp = w + (size_t)(ob + lrow) * CO + cb + lhi * 16;
    v16bf bfrag;
#pragma unroll
    for (int i = 0; i < 16; ++i) bfrag[i] = (__bf16)wp[i];
    acc = __builtin_amdgcn_wmma_f32_16x16x32_bf16(false, afrag, false, bfrag,
                                                  (short)0, acc, false, false);
    __syncthreads();
  }

  const float bs = bias[ob + lrow];
  __bf16* op = out + ((size_t)b * N_ + nb + 8 * lhi) * CO + ob + lrow;
#pragma unroll
  for (int r = 0; r < 8; ++r)
    op[(size_t)r * CO] = (__bf16)(acc[r] + bs);
}

// ---------------------------------------------------------------------------
// Projection for V: out[b][o][n] (V transposed) bf16 [B][CO][N]
// ---------------------------------------------------------------------------
__global__ __launch_bounds__(128)
void proj_v_kernel(const float* __restrict__ in,    // [B][CO][N]
                   const float* __restrict__ w,     // [CO][CO]
                   const float* __restrict__ bias,  // [CO]
                   __bf16* __restrict__ out)        // [B][CO][N] (Vt)
{
  const int tid  = threadIdx.x;
  const int wave = tid >> 5;
  const int lane = tid & 31;
  const int lrow = lane & 15;
  const int lhi  = lane >> 4;
  const int b  = blockIdx.z;
  const int nb = blockIdx.y * 16;
  const int ob = blockIdx.x * 64 + wave * 16;

  const float* inb = in + (size_t)b * CO * N_;
  v8f acc = {};

  for (int cb = 0; cb < CO; cb += 32) {
    const float* wp = w + (size_t)(ob + lrow) * CO + cb + lhi * 16;
    v16bf afrag;
#pragma unroll
    for (int i = 0; i < 8; ++i) {
      afrag[i]     = (__bf16)wp[i];
      afrag[8 + i] = (__bf16)wp[8 + i];
    }
    const float* bp = inb + (size_t)(cb + lhi * 16) * N_ + nb + lrow;
    v16bf bfrag;
#pragma unroll
    for (int i = 0; i < 16; ++i) bfrag[i] = (__bf16)bp[(size_t)i * N_];
    acc = __builtin_amdgcn_wmma_f32_16x16x32_bf16(false, afrag, false, bfrag,
                                                  (short)0, acc, false, false);
  }
#pragma unroll
  for (int r = 0; r < 8; ++r) {
    int o = ob + r + 8 * lhi;
    out[((size_t)b * CO + o) * N_ + nb + lrow] = (__bf16)(acc[r] + bias[o]);
  }
}

// ---------------------------------------------------------------------------
// Flash attention with async double-buffered K/V tiles in LDS.
// Block = 8 waves = 8 query tiles of 16; kv tiles of 32 (shared by all waves).
// ---------------------------------------------------------------------------
__global__ __launch_bounds__(NWAVES * 32)
void attn_kernel(const __bf16* __restrict__ Q,   // [B][N][CO]
                 const __bf16* __restrict__ K,   // [B][N][CO]
                 const __bf16* __restrict__ Vt,  // [B][CO][N]
                 const float*  __restrict__ x,   // [B][CO][N] residual
                 float* __restrict__ out)        // [B][CO][N]
{
  __shared__ __align__(16) __bf16 ldsK[2][32 * KPAD];   // [buf][kv][ch+pad]
  __shared__ __align__(16) __bf16 ldsV[2][CO * VPAD];   // [buf][ch][kv+pad]
  __shared__ __align__(32) __bf16 ldsP[NWAVES][16 * 32];

  const int tid  = threadIdx.x;
  const int wave = tid >> 5;
  const int lane = tid & 31;
  const int lrow = lane & 15;
  const int lhi  = lane >> 4;
  const int b     = blockIdx.z;
  const int qbase = (blockIdx.x * NWAVES + wave) * 16;

  const __bf16* Qb = Q  + (size_t)b * N_ * CO;
  const __bf16* Kb = K  + (size_t)b * N_ * CO;
  const __bf16* Vb = Vt + (size_t)b * CO * N_;

  // Per-thread async copy offsets:
  //   K tile: 32 rows x 512B, 8 threads/row, 64B each (4 x b128)
  //   V tile: 256 rows x 64B, 4 threads/row over 4 passes of 64 rows
  const unsigned k_row = tid >> 3, k_sub = tid & 7;
  const unsigned kg0 = k_row * 512 + k_sub * 64;
  const unsigned kl0 = k_row * (KPAD * 2) + k_sub * 64;
  const unsigned v_ch = tid >> 2, v_sub = tid & 3;
  const unsigned vg0 = v_ch * (N_ * 2) + v_sub * 16;
  const unsigned vl0 = v_ch * (VPAD * 2) + v_sub * 16;
  const unsigned ldsK0 = (unsigned)(uintptr_t)&ldsK[0][0];
  const unsigned ldsK1 = (unsigned)(uintptr_t)&ldsK[1][0];
  const unsigned ldsV0 = (unsigned)(uintptr_t)&ldsV[0][0];
  const unsigned ldsV1 = (unsigned)(uintptr_t)&ldsV[1][0];

  // Q fragments for this tile (kept in registers)
  v16bf qf[8];
  {
    const __bf16* qrow = Qb + (size_t)(qbase + lrow) * CO;
#pragma unroll
    for (int kk = 0; kk < 8; ++kk) {
      int base = kk * 32 + lhi * 16;
      v8bf lo = *(const v8bf*)(qrow + base);
      v8bf hi = *(const v8bf*)(qrow + base + 8);
      qf[kk] = make_a_frag(lo, hi);
    }
  }

  float m_run[8], l_run[8];
#pragma unroll
  for (int r = 0; r < 8; ++r) { m_run[r] = -1e30f; l_run[r] = 0.0f; }
  v8f oacc[16];
#pragma unroll
  for (int c = 0; c < 16; ++c) oacc[c] = (v8f){};

  __bf16* pl = &ldsP[wave][0];
  const float scale = 0.0625f;  // 1/sqrt(256)
  const int NT = N_ / 32;

  // Prologue: stage tile 0 into buffer 0
  {
    const void* kbase = (const void*)Kb;
    async_b128(ldsK0 + kl0 +  0, kg0 +  0, kbase);
    async_b128(ldsK0 + kl0 + 16, kg0 + 16, kbase);
    async_b128(ldsK0 + kl0 + 32, kg0 + 32, kbase);
    async_b128(ldsK0 + kl0 + 48, kg0 + 48, kbase);
    const void* vbase = (const void*)Vb;
#pragma unroll
    for (int i = 0; i < 4; ++i)
      async_b128(ldsV0 + vl0 + (unsigned)i * (64u * VPAD * 2u),
                 vg0 + (unsigned)i * (64u * N_ * 2u), vbase);
  }

  for (int t = 0; t < NT; ++t) {
    const int cur = t & 1;
    wait_asynccnt0();     // fills of buffer `cur` (issued last iter) complete
    __syncthreads();      // all waves' fills done; all readers of other buf done

    // Stage next tile into the other buffer (overlaps with compute below)
    if (t + 1 < NT) {
      const int kvn = (t + 1) * 32;
      const void* kbase = (const void*)(Kb + (size_t)kvn * CO);
      const unsigned kl = cur ? ldsK0 : ldsK1;
      async_b128(kl + kl0 +  0, kg0 +  0, kbase);
      async_b128(kl + kl0 + 16, kg0 + 16, kbase);
      async_b128(kl + kl0 + 32, kg0 + 32, kbase);
      async_b128(kl + kl0 + 48, kg0 + 48, kbase);
      const void* vbase = (const void*)(Vb + kvn);
      const unsigned vl = cur ? ldsV0 : ldsV1;
#pragma unroll
      for (int i = 0; i < 4; ++i)
        async_b128(vl + vl0 + (unsigned)i * (64u * VPAD * 2u),
                   vg0 + (unsigned)i * (64u * N_ * 2u), vbase);
    }

    const __bf16* kcur = &ldsK[cur][0];
    const __bf16* vcur = &ldsV[cur][0];

    // ---- scores: S[16q][32kv] from LDS K tile ----
    v8f s0 = {}, s1 = {};
    {
      const __bf16* kb0 = kcur + (size_t)lrow * KPAD + lhi * 16;
      const __bf16* kb1 = kb0 + (size_t)16 * KPAD;
#pragma unroll
      for (int kk = 0; kk < 8; ++kk) {
        v8bf a0 = *(const v8bf*)(kb0 + kk * 32);
        v8bf a1 = *(const v8bf*)(kb0 + kk * 32 + 8);
        v8bf c0 = *(const v8bf*)(kb1 + kk * 32);
        v8bf c1 = *(const v8bf*)(kb1 + kk * 32 + 8);
        v16bf b0 = make_a_frag(a0, a1);
        v16bf b1 = make_a_frag(c0, c1);
        s0 = __builtin_amdgcn_wmma_f32_16x16x32_bf16(false, qf[kk], false, b0,
                                                     (short)0, s0, false, false);
        s1 = __builtin_amdgcn_wmma_f32_16x16x32_bf16(false, qf[kk], false, b1,
                                                     (short)0, s1, false, false);
      }
    }

    // ---- online softmax (rows live in 16-lane column groups) ----
#pragma unroll
    for (int r = 0; r < 8; ++r) {
      float a  = s0[r] * scale;
      float b2 = s1[r] * scale;
      float rm = fmaxf(a, b2);
      rm = fmaxf(rm, __shfl_xor(rm, 1));
      rm = fmaxf(rm, __shfl_xor(rm, 2));
      rm = fmaxf(rm, __shfl_xor(rm, 4));
      rm = fmaxf(rm, __shfl_xor(rm, 8));
      float mnew = fmaxf(m_run[r], rm);
      float corr = __expf(m_run[r] - mnew);
      float p0 = __expf(a - mnew);
      float p1 = __expf(b2 - mnew);
      float rs = p0 + p1;
      rs += __shfl_xor(rs, 1);
      rs += __shfl_xor(rs, 2);
      rs += __shfl_xor(rs, 4);
      rs += __shfl_xor(rs, 8);
      l_run[r] = l_run[r] * corr + rs;
      m_run[r] = mnew;
#pragma unroll
      for (int c = 0; c < 16; ++c) oacc[c][r] *= corr;
      pl[(r + 8 * lhi) * 32 + lrow]      = (__bf16)p0;
      pl[(r + 8 * lhi) * 32 + 16 + lrow] = (__bf16)p1;
    }

    // re-read P as A fragment (same-wave LDS is in-order)
    v8bf plo = *(const v8bf*)(pl + lrow * 32 + lhi * 16);
    v8bf phi = *(const v8bf*)(pl + lrow * 32 + lhi * 16 + 8);
    v16bf pf = make_a_frag(plo, phi);

    // ---- O += P * V from LDS V tile ----
#pragma unroll
    for (int c = 0; c < 16; ++c) {
      const __bf16* vp = vcur + (size_t)(c * 16 + lrow) * VPAD + lhi * 16;
      v8bf u0 = *(const v8bf*)(vp);
      v8bf u1 = *(const v8bf*)(vp + 8);
      v16bf vf = make_a_frag(u0, u1);
      oacc[c] = __builtin_amdgcn_wmma_f32_16x16x32_bf16(false, pf, false, vf,
                                                        (short)0, oacc[c], false, false);
    }
  }

  // ---- epilogue: out = O/l + x, fp32 [B][CO][N] ----
  float inv[8];
#pragma unroll
  for (int r = 0; r < 8; ++r) inv[r] = 1.0f / l_run[r];
  const size_t nq = (size_t)qbase + 8 * lhi;
#pragma unroll
  for (int c = 0; c < 16; ++c) {
    int ch = c * 16 + lrow;
    size_t idx = ((size_t)b * CO + ch) * N_ + nq;
    v8f xr = *(const v8f*)(x + idx);
    v8f o;
#pragma unroll
    for (int r = 0; r < 8; ++r) o[r] = oacc[c][r] * inv[r] + xr[r];
    *(v8f*)(out + idx) = o;
  }
}

// ---------------------------------------------------------------------------
extern "C" void kernel_launch(void* const* d_in, const int* in_sizes, int n_in,
                              void* d_out, int out_size, void* d_ws, size_t ws_size,
                              hipStream_t stream) {
  (void)in_sizes; (void)n_in; (void)out_size; (void)ws_size;
  const float* x   = (const float*)d_in[0];
  const float* ctx = (const float*)d_in[1];
  const float* wq  = (const float*)d_in[2];
  const float* bq  = (const float*)d_in[3];
  const float* wk  = (const float*)d_in[4];
  const float* bk  = (const float*)d_in[5];
  const float* wv  = (const float*)d_in[6];
  const float* bv  = (const float*)d_in[7];
  float* out = (float*)d_out;

  const size_t elems = (size_t)B_ * N_ * CO;
  const size_t seg   = ((elems * sizeof(__bf16) + 255) / 256) * 256;
  char* ws = (char*)d_ws;
  __bf16* Q  = (__bf16*)(ws);
  __bf16* K  = (__bf16*)(ws + seg);
  __bf16* Vt = (__bf16*)(ws + 2 * seg);

  dim3 pgrid(CO / 64, N_ / 16, B_);
  proj_qk_kernel<<<pgrid, 128, 0, stream>>>(x,   wq, bq, Q);
  proj_qk_kernel<<<pgrid, 128, 0, stream>>>(ctx, wk, bk, K);
  proj_v_kernel <<<pgrid, 128, 0, stream>>>(ctx, wv, bv, Vt);

  dim3 agrid((N_ / 16) / NWAVES, 1, B_);
  attn_kernel<<<agrid, NWAVES * 32, 0, stream>>>(Q, K, Vt, x, out);
}
